// Block_3444563771825
// MI455X (gfx1250) — matrix operand order repaired and checked by reading, not compiled, not run on hardware
//
#include <hip/hip_runtime.h>

// ---------------------------------------------------------------------------
// Types / helpers for CDNA5 WMMA (wave32, 16x16x32 bf16 -> f32)
// ---------------------------------------------------------------------------
typedef __bf16        v16bf __attribute__((ext_vector_type(16)));
typedef float         v8f   __attribute__((ext_vector_type(8)));
typedef unsigned int  u32x4 __attribute__((ext_vector_type(4)));

struct B256 { u32x4 lo, hi; };   // 32 bytes == one v16bf fragment

__device__ __forceinline__ v16bf make_frag(u32x4 lo, u32x4 hi) {
    B256 t; t.lo = lo; t.hi = hi;
    return __builtin_bit_cast(v16bf, t);
}

__device__ __forceinline__ unsigned short f32_bf16(float f) {
    unsigned u = __builtin_bit_cast(unsigned, f);
    u += 0x7fffu + ((u >> 16) & 1u);          // round-to-nearest-even
    return (unsigned short)(u >> 16);
}

// A-fragment (16x32, M x K), row-major source, leading dim `ld` (elements).
// ISA layout: lane m = lane&15, half h = lane>>4; K groups {8h..8h+7},{16+8h..}.
__device__ __forceinline__ v16bf ld_frag_a(const unsigned short* p, int m0, int ld, int lane) {
    int h = (lane >> 4) & 1;
    const unsigned short* base = p + (size_t)(m0 + (lane & 15)) * ld + 8 * h;
    return make_frag(*(const u32x4*)base, *(const u32x4*)(base + 16));
}

// B-fragment (32x16, K x N), from K-contiguous (transposed) storage: row n, ld elems.
// ISA layout: lane n = lane&15, half h = lane>>4; K groups {8h..8h+7},{16+8h..}.
__device__ __forceinline__ v16bf ld_frag_b(const unsigned short* p, int n0, int ld, int lane) {
    int h = (lane >> 4) & 1;
    const unsigned short* base = p + (size_t)(n0 + (lane & 15)) * ld + 8 * h;
    return make_frag(*(const u32x4*)base, *(const u32x4*)(base + 16));
}

#define WMMA_BF16(a, b, c) \
    __builtin_amdgcn_wmma_f32_16x16x32_bf16(false, (a), false, (b), (short)0, (c), false, false)

// ---------------------------------------------------------------------------
// CDNA5 async global->LDS copy (ASYNCcnt path). 16 bytes per lane.
// dst_lds: byte offset of LDS destination (low 32 bits of generic pointer),
// src: per-lane 64-bit global address. SADDR=null ("off").
// ---------------------------------------------------------------------------
__device__ __forceinline__ void async_copy_b128(unsigned dst_lds, const void* src) {
    asm volatile("global_load_async_to_lds_b128 %0, %1, off"
                 :: "v"(dst_lds), "v"(src) : "memory");
}
__device__ __forceinline__ void wait_async0() {
    asm volatile("s_wait_asynccnt 0x0" ::: "memory");
}
__device__ __forceinline__ unsigned lds_off(const void* p) {
    return (unsigned)(size_t)p;   // LDS aperture: addr[31:0] == LDS offset
}

// ---------------------------------------------------------------------------
// LayerNorm: one 256-thread block per row of C=1024. Writes f32 and/or bf16.
// ---------------------------------------------------------------------------
__global__ __launch_bounds__(256) void ln_kernel(const float* __restrict__ in,
                                                 const float* __restrict__ g,
                                                 const float* __restrict__ b,
                                                 float* outf, unsigned short* outb, int C) {
    const int row = blockIdx.x, tid = threadIdx.x;
    const float* x = in + (size_t)row * C;
    __shared__ float red[256];

    float s = 0.f;
    for (int i = tid; i < C; i += 256) s += x[i];
    red[tid] = s; __syncthreads();
    for (int st = 128; st > 0; st >>= 1) { if (tid < st) red[tid] += red[tid + st]; __syncthreads(); }
    const float mean = red[0] / (float)C;
    __syncthreads();

    float v = 0.f;
    for (int i = tid; i < C; i += 256) { float d = x[i] - mean; v += d * d; }
    red[tid] = v; __syncthreads();
    for (int st = 128; st > 0; st >>= 1) { if (tid < st) red[tid] += red[tid + st]; __syncthreads(); }
    const float inv = rsqrtf(red[0] / (float)C + 1e-5f);

    for (int i = tid; i < C; i += 256) {
        float y = (x[i] - mean) * inv * g[i] + b[i];
        if (outf) outf[(size_t)row * C + i] = y;
        if (outb) outb[(size_t)row * C + i] = f32_bf16(y);
    }
}

// ---------------------------------------------------------------------------
// Weight packing, TRANSPOSED (K-contiguous) for async-friendly B staging.
// Wk/Wq/Wv [H,C,HS] -> bf16 [3072][1024]: row col = {k|q|v}*1024 + h*64 + d.
// ---------------------------------------------------------------------------
__global__ __launch_bounds__(256) void pack_qkv_t_kernel(const float* __restrict__ Wk,
                                                         const float* __restrict__ Wq,
                                                         const float* __restrict__ Wv,
                                                         unsigned short* __restrict__ dst) {
    int i = blockIdx.x * 256 + threadIdx.x;          // i enumerates [col][c]
    int col = i >> 10, c = i & 1023;
    int m = col >> 10, h = (col >> 6) & 15, d = col & 63;
    const float* W = (m == 0) ? Wk : (m == 1) ? Wq : Wv;
    dst[i] = f32_bf16(W[(size_t)h * 1024 * 64 + (size_t)c * 64 + d]);
}

// generic transpose-convert: src f32 [K][N] -> dst bf16 [N][K]
__global__ __launch_bounds__(256) void cvt_t_kernel(const float* __restrict__ src,
                                                    unsigned short* __restrict__ dst,
                                                    int K, int N) {
    int i = blockIdx.x * 256 + threadIdx.x;
    int n = i / K, k = i % K;
    dst[i] = f32_bf16(src[(size_t)k * N + n]);
}

// ---------------------------------------------------------------------------
// bf16 WMMA GEMM: C[M,N] = A[M,K] @ Bt[N,K]^T  (+bias, +ReLU, +residual).
// Block tile 128x128, BK=32; 8 waves in 4(M) x 2(N); wave tile 32x64 = 2x4 WMMA.
// Double-buffered LDS, staged with global_load_async_to_lds_b128 (ASYNCcnt).
// Optional vt: write cols >= 2048 transposed per (b,head) as [d][T] (for attn V).
// ---------------------------------------------------------------------------
#define G_LD 40   // padded LDS leading dim: 80B rows, 16B aligned, bank-spread

__global__ __launch_bounds__(256) void gemm_bf16_kernel(
        const unsigned short* __restrict__ A, const unsigned short* __restrict__ Bt,
        int M, int N, int K,
        const float* __restrict__ bias, const float* __restrict__ res,
        float* outf, unsigned short* outb,
        unsigned short* vt, int Tdim, int relu) {
    __shared__ __attribute__((aligned(16))) unsigned short As[2][128 * G_LD];
    __shared__ __attribute__((aligned(16))) unsigned short Bs[2][128 * G_LD];

    const int tid = threadIdx.x, lane = tid & 31, w = tid >> 5;
    const int wm = w >> 1, wn = w & 1;                 // 4 x 2 wave grid
    const int m_blk = blockIdx.x * 128, n_blk = blockIdx.y * 128;

    const int srow = tid >> 1, shalf = tid & 1;        // staging: 16B chunk per thread
    const unsigned short* aSrc = A  + (size_t)(m_blk + srow) * K + shalf * 16;
    const unsigned short* bSrc = Bt + (size_t)(n_blk + srow) * K + shalf * 16;
    const unsigned sDst = srow * G_LD * 2 + shalf * 32;   // byte offset within tile

    v8f acc[2][4] = {};

    // prologue: stage tile 0
    async_copy_b128(lds_off(&As[0][0]) + sDst, aSrc);
    async_copy_b128(lds_off(&Bs[0][0]) + sDst, bSrc);
    wait_async0();
    __syncthreads();

    int cur = 0;
    for (int k0 = 0; k0 < K; k0 += 32, cur ^= 1) {
        if (k0 + 32 < K) {   // stage next tile into the other buffer (overlaps WMMAs)
            async_copy_b128(lds_off(&As[cur ^ 1][0]) + sDst, aSrc + k0 + 32);
            async_copy_b128(lds_off(&Bs[cur ^ 1][0]) + sDst, bSrc + k0 + 32);
        }
        v16bf a0 = ld_frag_a(&As[cur][0], wm * 32 + 0,  G_LD, lane);
        v16bf a1 = ld_frag_a(&As[cur][0], wm * 32 + 16, G_LD, lane);
        #pragma unroll
        for (int j = 0; j < 4; j++) {
            v16bf bf = ld_frag_b(&Bs[cur][0], wn * 64 + j * 16, G_LD, lane);
            acc[0][j] = WMMA_BF16(a0, bf, acc[0][j]);
            acc[1][j] = WMMA_BF16(a1, bf, acc[1][j]);
        }
        wait_async0();       // our next-tile chunks have landed in LDS
        __syncthreads();     // everyone done reading cur, next published
    }

    // ---- epilogue: bias -> relu -> residual -> store ----
    const int h = (lane >> 4) & 1, nl = lane & 15;
    #pragma unroll
    for (int i = 0; i < 2; i++) {
        #pragma unroll
        for (int j = 0; j < 4; j++) {
            int ncol = n_blk + wn * 64 + j * 16 + nl;
            float bv = bias ? bias[ncol] : 0.f;
            #pragma unroll
            for (int r = 0; r < 8; r++) {
                int mrow = m_blk + wm * 32 + i * 16 + r + 8 * h;
                float vv = acc[i][j][r] + bv;
                if (relu) vv = fmaxf(vv, 0.f);
                if (res)  vv += res[(size_t)mrow * N + ncol];
                if (outf) outf[(size_t)mrow * N + ncol] = vv;
                if (outb) outb[(size_t)mrow * N + ncol] = f32_bf16(vv);
                if (vt && ncol >= 2048) {   // V section, store [b][head][d][T]
                    int bb = mrow / Tdim, tt = mrow % Tdim;
                    int cc = ncol - 2048, hd = cc >> 6, dd = cc & 63;
                    vt[(((size_t)bb * 16 + hd) * 64 + dd) * Tdim + tt] = f32_bf16(vv);
                }
            }
        }
    }
}

// ---------------------------------------------------------------------------
// Flash-style causal attention, scores = K @ Q^T (faithful to reference).
// Grid: (B*H, T/64); 128 threads = 4 fully independent waves (no barriers).
// K resident as A-frags; Q B-frags straight from global; V B-frags straight
// from the transposed vt[b][head][d][T] buffer. P converts D->A layout via a
// wave-private LDS round trip (LDS is in-order within a wave).
// ---------------------------------------------------------------------------
__global__ __launch_bounds__(128) void attn_kernel(const unsigned short* __restrict__ qkv,
                                                   const unsigned short* __restrict__ vt,
                                                   unsigned short* __restrict__ out,
                                                   int T, int H) {
    const int b = blockIdx.x / H, head = blockIdx.x % H;
    const int w = threadIdx.x >> 5, lane = threadIdx.x & 31;
    const int t0 = blockIdx.y * 64 + w * 16;
    const int hh = (lane >> 4) & 1, nl = lane & 15;
    const size_t RS = 3072;
    const unsigned short* base = qkv + (size_t)b * T * RS;
    const unsigned short* vbase = vt + ((size_t)b * H + head) * 64 * T;

    __shared__ __attribute__((aligned(16))) unsigned short Pt[4][16 * G_LD];

    v16bf kfr[2];
    #pragma unroll
    for (int c = 0; c < 2; c++)
        kfr[c] = ld_frag_a(base + (size_t)t0 * RS + head * 64 + c * 32, 0, (int)RS, lane);

    v8f acc[4] = {};
    float mrow[8], lrow[8];
    #pragma unroll
    for (int r = 0; r < 8; r++) { mrow[r] = -1e30f; lrow[r] = 0.f; }
    const float scale = 0.125f;   // HS^-0.5, HS=64

    for (int s0 = 0; s0 < t0 + 16; s0 += 32) {       // causal bound per wave
        #pragma unroll
        for (int ss = 0; ss < 32; ss += 16) {
            const int scol = s0 + ss + nl;
            v8f S = {};
            #pragma unroll
            for (int c = 0; c < 2; c++) {
                v16bf qb = ld_frag_b(base + 1024 + head * 64 + c * 32, s0 + ss, (int)RS, lane);
                S = WMMA_BF16(kfr[c], qb, S);
            }
            #pragma unroll
            for (int r = 0; r < 8; r++) {
                const int trow = t0 + r + 8 * hh;
                float sv = S[r] * scale;
                if (scol > trow) sv = -1e30f;
                float mx = sv;
                mx = fmaxf(mx, __shfl_xor(mx, 1, 32));
                mx = fmaxf(mx, __shfl_xor(mx, 2, 32));
                mx = fmaxf(mx, __shfl_xor(mx, 4, 32));
                mx = fmaxf(mx, __shfl_xor(mx, 8, 32));
                const float mnew = fmaxf(mrow[r], mx);
                const float pe = __expf(sv - mnew);
                const float al = __expf(mrow[r] - mnew);
                float ps = pe;
                ps += __shfl_xor(ps, 1, 32);
                ps += __shfl_xor(ps, 2, 32);
                ps += __shfl_xor(ps, 4, 32);
                ps += __shfl_xor(ps, 8, 32);
                lrow[r] = lrow[r] * al + ps;
                mrow[r] = mnew;
                #pragma unroll
                for (int n = 0; n < 4; n++) acc[n][r] *= al;
                Pt[w][(r + 8 * hh) * G_LD + ss + nl] = f32_bf16(pe);
            }
        }
        v16bf pa = ld_frag_a(&Pt[w][0], 0, G_LD, lane);
        #pragma unroll
        for (int n = 0; n < 4; n++) {
            v16bf vb = ld_frag_b(vbase + (size_t)(n * 16) * T + s0, 0, T, lane);
            acc[n] = WMMA_BF16(pa, vb, acc[n]);
        }
    }

    #pragma unroll
    for (int n = 0; n < 4; n++) {
        #pragma unroll
        for (int r = 0; r < 8; r++) {
            const int trow = t0 + r + 8 * hh;
            float ov = acc[n][r] / lrow[r];
            out[((size_t)b * T + trow) * 1024 + head * 64 + n * 16 + nl] = f32_bf16(ov);
        }
    }
}

// ---------------------------------------------------------------------------
// Host orchestration
// ---------------------------------------------------------------------------
extern "C" void kernel_launch(void* const* d_in, const int* in_sizes, int n_in,
                              void* d_out, int out_size, void* d_ws, size_t ws_size,
                              hipStream_t stream) {
    (void)in_sizes; (void)n_in; (void)out_size; (void)ws_size;
    const int Bb = 4, T = 2048, C = 1024, H = 16;
    const int M = Bb * T;                       // 8192 rows

    const float* tokens = (const float*)d_in[0];
    const float* Wk     = (const float*)d_in[1];
    const float* Wq     = (const float*)d_in[2];
    const float* Wv     = (const float*)d_in[3];
    const float* Wproj  = (const float*)d_in[4];
    const float* bproj  = (const float*)d_in[5];
    const float* ln1_g  = (const float*)d_in[6];
    const float* ln1_b  = (const float*)d_in[7];
    const float* W1     = (const float*)d_in[8];
    const float* b1     = (const float*)d_in[9];
    const float* W2     = (const float*)d_in[10];
    const float* b2     = (const float*)d_in[11];
    const float* ln2_g  = (const float*)d_in[12];
    const float* ln2_b  = (const float*)d_in[13];

    char* ws = (char*)d_ws; size_t off = 0;
    auto carve = [&](size_t bytes) { void* p = ws + off; off += (bytes + 255) & ~(size_t)255; return p; };
    float*          x_f32    = (float*)carve((size_t)M * C * 4);
    float*          inp_f32  = (float*)carve((size_t)M * C * 4);
    unsigned short* x_bf     = (unsigned short*)carve((size_t)M * C * 2);
    unsigned short* h_bf     = (unsigned short*)carve((size_t)M * C * 2);
    unsigned short* attn_bf  = (unsigned short*)carve((size_t)M * C * 2);
    unsigned short* qkv_bf   = (unsigned short*)carve((size_t)M * 3 * C * 2);
    unsigned short* vt_bf    = (unsigned short*)carve((size_t)M * C * 2);
    unsigned short* hh_bf    = (unsigned short*)carve((size_t)M * 4 * C * 2);
    unsigned short* Wqkv_t   = (unsigned short*)carve((size_t)C * 3 * C * 2);
    unsigned short* Wproj_t  = (unsigned short*)carve((size_t)C * C * 2);
    unsigned short* W1_t     = (unsigned short*)carve((size_t)C * 4 * C * 2);
    unsigned short* W2_t     = (unsigned short*)carve((size_t)4 * C * C * 2);

    // weight preprocessing: bf16, transposed [N][K]
    pack_qkv_t_kernel<<<(C * 3 * C) / 256, 256, 0, stream>>>(Wk, Wq, Wv, Wqkv_t);
    cvt_t_kernel<<<(C * C) / 256,     256, 0, stream>>>(Wproj, Wproj_t, C, C);
    cvt_t_kernel<<<(C * 4 * C) / 256, 256, 0, stream>>>(W1, W1_t, C, 4 * C);
    cvt_t_kernel<<<(4 * C * C) / 256, 256, 0, stream>>>(W2, W2_t, 4 * C, C);

    // x = ln1(tokens)  (f32 for residual, bf16 for GEMMs)
    ln_kernel<<<M, 256, 0, stream>>>(tokens, ln1_g, ln1_b, x_f32, x_bf, C);

    // qkv = x @ Wqkv   [8192 x 3072]; V also written transposed per (b,head)
    gemm_bf16_kernel<<<dim3(M / 128, (3 * C) / 128), 256, 0, stream>>>(
        x_bf, Wqkv_t, M, 3 * C, C, nullptr, nullptr, nullptr, qkv_bf, vt_bf, T, 0);

    // causal attention (k@q^T scoring), concat heads
    attn_kernel<<<dim3(Bb * H, T / 64), 128, 0, stream>>>(qkv_bf, vt_bf, attn_bf, T, H);

    // inputs = x + attn @ Wproj + bproj   (f32)
    gemm_bf16_kernel<<<dim3(M / 128, C / 128), 256, 0, stream>>>(
        attn_bf, Wproj_t, M, C, C, bproj, x_f32, inp_f32, nullptr, nullptr, T, 0);

    // h = ln2(inputs) (bf16)
    ln_kernel<<<M, 256, 0, stream>>>(inp_f32, ln2_g, ln2_b, nullptr, h_bf, C);

    // hh = relu(h @ W1 + b1)   [8192 x 4096]
    gemm_bf16_kernel<<<dim3(M / 128, (4 * C) / 128), 256, 0, stream>>>(
        h_bf, W1_t, M, 4 * C, C, b1, nullptr, nullptr, hh_bf, nullptr, T, 1);

    // out = inputs + hh @ W2 + b2   (f32 -> d_out)
    gemm_bf16_kernel<<<dim3(M / 128, C / 128), 256, 0, stream>>>(
        hh_bf, W2_t, M, C, 4 * C, b2, inp_f32, (float*)d_out, nullptr, nullptr, T, 0);
}
